// SelfAttention_26560077758891
// MI455X (gfx1250) — matrix operand order repaired and checked by reading, not compiled, not run on hardware
//
#include <hip/hip_runtime.h>
#include <hip/hip_bf16.h>
#include <stdint.h>

// ---------------------------------------------------------------------------
// MI455X / gfx1250 self-attention, bf16 WMMA everywhere (fp32 accumulate).
//   B=4, T=2048, C=1024, H=16, D=64
// ---------------------------------------------------------------------------

typedef __bf16 bf16_t;
typedef __attribute__((ext_vector_type(16))) __bf16 v16bf;
typedef __attribute__((ext_vector_type(8)))  __bf16 v8bf;
typedef __attribute__((ext_vector_type(4)))  __bf16 v4bf;
typedef __attribute__((ext_vector_type(8)))  float  v8f;
typedef __attribute__((ext_vector_type(4)))  float  v4f;

#define BATCH   4
#define T_SEQ   2048
#define CDIM    1024
#define HEADS   16
#define DHEAD   64

// ---- WMMA wrapper: v_wmma_f32_16x16x32_bf16 -------------------------------
__device__ __forceinline__ v8f wmma_bf16(v16bf a, v16bf b, v8f c) {
  return __builtin_amdgcn_wmma_f32_16x16x32_bf16(
      /*neg_a=*/false, a, /*neg_b=*/false, b,
      /*c_mod=*/(short)0, c, /*reuse_a=*/false, /*reuse_b=*/false);
}

__device__ __forceinline__ v16bf cat8(v8bf lo, v8bf hi) {
  v16bf f;
#pragma unroll
  for (int i = 0; i < 8; ++i) { f[i] = lo[i]; f[i + 8] = hi[i]; }
  return f;
}

// A fragment 16(M)x32(K) bf16 (ISA 7.12.2):
//   lane<16 : row=lane,    halves[0..7]=K0..7,  halves[8..15]=K16..23
//   lane>=16: row=lane-16, halves[0..7]=K8..15, halves[8..15]=K24..31
__device__ __forceinline__ v16bf load_a_frag(const bf16_t* base, int ldHalfs) {
  const int lane = threadIdx.x & 31;
  const int row  = lane & 15;
  const int koff = (lane >> 4) * 8;
  const bf16_t* p = base + (size_t)row * ldHalfs + koff;
  return cat8(*(const v8bf*)(p), *(const v8bf*)(p + 16));
}

// B fragment 32(K)x16(N) bf16 from COLUMN-major storage bt[n][k]:
//   lane<16 : col=lane,    halves[0..7]=K0..7,  halves[8..15]=K16..23
//   lane>=16: col=lane-16, halves[0..7]=K8..15, halves[8..15]=K24..31
__device__ __forceinline__ v16bf load_b_frag_cm(const bf16_t* bt, int ldHalfs) {
  const int lane = threadIdx.x & 31;
  const int col  = lane & 15;
  const int koff = (lane >> 4) * 8;
  const bf16_t* p = bt + (size_t)col * ldHalfs + koff;
  return cat8(*(const v8bf*)(p), *(const v8bf*)(p + 16));
}

// ---------------------------------------------------------------------------
// Kernel 1: KQV GEMM.  X(8192x1024 f32) * Wkqv(1024x3072 f32) -> bf16,
// scattered into K/Q/V [B][H][T][D].
// WG tile: 128(M)x128(N); 8 waves as 4(m)x2(n); wave tile 32x64 = 2x4 WMMA.
// ---------------------------------------------------------------------------
__global__ __launch_bounds__(256)
void qkv_gemm_kernel(const float* __restrict__ X, const float* __restrict__ W,
                     bf16_t* __restrict__ Kb, bf16_t* __restrict__ Qb,
                     bf16_t* __restrict__ Vb)
{
  __shared__ __align__(16) bf16_t Alds[128 * 40];  // [m][k], stride 40 (80B)
  __shared__ __align__(16) bf16_t Bt[128 * 40];    // [n][k] transposed

  const int m0   = blockIdx.x * 128;
  const int n0   = blockIdx.y * 128;
  const int tid  = threadIdx.x;
  const int wave = tid >> 5;
  const int wm   = (wave >> 1) * 32;               // 4 waves down
  const int wn   = (wave & 1) * 64;                // 2 waves across

  v8f acc[2][4] = {};

  for (int k0 = 0; k0 < CDIM; k0 += 32) {
    __syncthreads();
    // Stage A: 128x32 f32 -> bf16, 4 float4 per thread.
#pragma unroll
    for (int i = 0; i < 4; ++i) {
      int c = tid + i * 256;                       // 1024 float4 chunks
      int row = c >> 3, col4 = (c & 7) * 4;
      v4f xv = *(const v4f*)(X + (size_t)(m0 + row) * CDIM + k0 + col4);
      v4bf bv;
#pragma unroll
      for (int j = 0; j < 4; ++j) bv[j] = (bf16_t)xv[j];
      *(v4bf*)(Alds + row * 40 + col4) = bv;
    }
    // Stage B transposed: 32x128 f32 -> bf16 [n][k], 4 float4 per thread.
#pragma unroll
    for (int i = 0; i < 4; ++i) {
      int c = tid + i * 256;                       // 1024 float4 chunks
      int krow = c >> 5, col4 = (c & 31) * 4;
      v4f wv = *(const v4f*)(W + (size_t)(k0 + krow) * (3 * CDIM) + n0 + col4);
#pragma unroll
      for (int j = 0; j < 4; ++j) Bt[(col4 + j) * 40 + krow] = (bf16_t)wv[j];
    }
    // Prefetch next K-tile while this one computes (global_prefetch_b8).
    if (k0 + 32 < CDIM) {
      __builtin_prefetch(X + (size_t)(m0 + (tid >> 3)) * CDIM + k0 + 32 + (tid & 7) * 4, 0, 0);
      __builtin_prefetch(W + (size_t)(k0 + 32 + (tid >> 5)) * (3 * CDIM) + n0 + (tid & 31) * 4, 0, 0);
    }
    __syncthreads();

#pragma unroll
    for (int mi = 0; mi < 2; ++mi) {
      v16bf a = load_a_frag(Alds + (wm + mi * 16) * 40, 40);
#pragma unroll
      for (int ni = 0; ni < 4; ++ni) {
        v16bf b = load_b_frag_cm(Bt + (wn + ni * 16) * 40, 40);
        acc[mi][ni] = wmma_bf16(a, b, acc[mi][ni]);
      }
    }
  }

  // Scatter C-layout tiles into K/Q/V[b][h][t][d] (bf16).
  // n0 is a multiple of 128 and 1024 % 128 == 0 -> `which` is WG-constant;
  // each 16-wide WMMA tile stays inside one 64-wide head block.
  const int lane  = tid & 31;
  const int col   = lane & 15;
  const int rbase = (lane >> 4) * 8;
  const int which = n0 >> 10;                      // 0=K, 1=Q, 2=V (k first!)
  bf16_t* dst = (which == 0) ? Kb : ((which == 1) ? Qb : Vb);

#pragma unroll
  for (int mi = 0; mi < 2; ++mi)
#pragma unroll
    for (int ni = 0; ni < 4; ++ni) {
      int noff = (n0 & (CDIM - 1)) + wn + ni * 16; // offset inside 1024 block
      int h  = noff >> 6;
      int d0 = noff & 63;
#pragma unroll
      for (int r = 0; r < 8; ++r) {
        int m = m0 + wm + mi * 16 + rbase + r;     // global row = b*T + t
        int b = m >> 11, t = m & (T_SEQ - 1);
        dst[(((size_t)(b * HEADS + h)) * T_SEQ + t) * DHEAD + d0 + col] =
            (bf16_t)acc[mi][ni][r];
      }
    }
}

// ---------------------------------------------------------------------------
// Kernel 2: fused flash attention for one (b,h), 128 q-rows per WG.
// Each wave: 16 q-rows, full D=64 accumulator (4 v8f WMMA tiles).
// Per 32-key tile: S = Q*K^T (4 WMMA), online softmax, O += P*V (4 WMMA).
// ---------------------------------------------------------------------------
__global__ __launch_bounds__(256)
void attn_kernel(const bf16_t* __restrict__ Qb, const bf16_t* __restrict__ Kb,
                 const bf16_t* __restrict__ Vb, bf16_t* __restrict__ Ob)
{
  __shared__ __align__(16) bf16_t Klds[32 * 72];       // [t][d], stride 72
  __shared__ __align__(16) bf16_t Vt[64 * 40];         // [d][t] transposed
  __shared__ __align__(16) bf16_t Plds[8][16 * 40];    // per-wave P tile

  const int h = blockIdx.y;
  const int b = blockIdx.z;
  const size_t headBase = ((size_t)(b * HEADS + h)) * T_SEQ * DHEAD;
  const bf16_t* Qh = Qb + headBase;
  const bf16_t* Kh = Kb + headBase;
  const bf16_t* Vh = Vb + headBase;

  const int tid  = threadIdx.x;
  const int wave = tid >> 5;
  const int lane = tid & 31;
  const int q0   = blockIdx.x * 128 + wave * 16;

  // Persistent Q A-fragments (d 0..31 and 32..63), straight from global.
  v16bf qa0 = load_a_frag(Qh + (size_t)q0 * DHEAD + 0, DHEAD);
  v16bf qa1 = load_a_frag(Qh + (size_t)q0 * DHEAD + 32, DHEAD);

  v8f o[4] = {};
  float mrow[8], lrow[8];
#pragma unroll
  for (int r = 0; r < 8; ++r) { mrow[r] = -1e30f; lrow[r] = 0.0f; }
  const float dk = 0.125f;                             // D^-0.5, D=64

  for (int kt = 0; kt < T_SEQ; kt += 32) {
    __syncthreads();
    {  // Stage K naturally and V transposed: one v8bf per thread each.
      int row = tid >> 3, chunk = (tid & 7) * 8;
      *(v8bf*)(Klds + row * 72 + chunk) =
          *(const v8bf*)(Kh + (size_t)(kt + row) * DHEAD + chunk);
      v8bf vv = *(const v8bf*)(Vh + (size_t)(kt + row) * DHEAD + chunk);
#pragma unroll
      for (int j = 0; j < 8; ++j) Vt[(chunk + j) * 40 + row] = vv[j];
      // Prefetch next key tile (global_prefetch_b8).
      if (kt + 32 < T_SEQ) {
        __builtin_prefetch(Kh + (size_t)(kt + 32 + row) * DHEAD + chunk, 0, 0);
        __builtin_prefetch(Vh + (size_t)(kt + 32 + row) * DHEAD + chunk, 0, 0);
      }
    }
    __syncthreads();

    // S(16q x 32k) = Q(16x64) * K^T(64x32): Klds is already column-major
    // for this B operand ([k-col][d-K]).
    v8f s0 = {}, s1 = {};
    s0 = wmma_bf16(qa0, load_b_frag_cm(Klds + 0,            72), s0);
    s0 = wmma_bf16(qa1, load_b_frag_cm(Klds + 32,           72), s0);
    s1 = wmma_bf16(qa0, load_b_frag_cm(Klds + 16 * 72,      72), s1);
    s1 = wmma_bf16(qa1, load_b_frag_cm(Klds + 16 * 72 + 32, 72), s1);

    // Online softmax. C-layout: lane groups of 16 share each row; reduce
    // with shfl_xor masks 1,2,4,8 (stay inside the 16-lane group).
    float alpha[8];
#pragma unroll
    for (int r = 0; r < 8; ++r) {
      float v0 = s0[r] * dk, v1 = s1[r] * dk;
      float mx = fmaxf(v0, v1);
#pragma unroll
      for (int msk = 1; msk < 16; msk <<= 1)
        mx = fmaxf(mx, __shfl_xor(mx, msk, 32));
      float mnew = fmaxf(mrow[r], mx);
      float a  = __expf(mrow[r] - mnew);
      float p0 = __expf(v0 - mnew);
      float p1 = __expf(v1 - mnew);
      float ps = p0 + p1;
#pragma unroll
      for (int msk = 1; msk < 16; msk <<= 1)
        ps += __shfl_xor(ps, msk, 32);
      lrow[r] = lrow[r] * a + ps;
      mrow[r] = mnew;
      alpha[r] = a;
      s0[r] = p0; s1[r] = p1;
    }
#pragma unroll
    for (int dt = 0; dt < 4; ++dt)
#pragma unroll
      for (int r = 0; r < 8; ++r) o[dt][r] *= alpha[r];

    // C-layout -> A-layout for P via per-wave LDS bounce (bf16 convert).
    bf16_t* pb = &Plds[wave][0];
    {
      int colc = lane & 15, rb = (lane >> 4) * 8;
#pragma unroll
      for (int r = 0; r < 8; ++r) {
        pb[(rb + r) * 40 + colc]      = (bf16_t)s0[r];
        pb[(rb + r) * 40 + 16 + colc] = (bf16_t)s1[r];
      }
    }
    v16bf pfrag = load_a_frag(pb, 40);     // per-wave RAW: LDS is in-order

    // O(16x64) += P(16x32) * V(32x64); Vt gives contiguous B columns.
#pragma unroll
    for (int dt = 0; dt < 4; ++dt) {
      v16bf bv = load_b_frag_cm(Vt + dt * 16 * 40, 40);
      o[dt] = wmma_bf16(pfrag, bv, o[dt]);
    }
  }

  // Normalize and store O as bf16 in [B][T][C] (head-concat) for proj GEMM.
  {
    const int colc = lane & 15, rb = (lane >> 4) * 8;
#pragma unroll
    for (int r = 0; r < 8; ++r) {
      float inv = 1.0f / lrow[r];
      int t = q0 + rb + r;
      size_t rowbase = ((size_t)b * T_SEQ + t) * CDIM + h * DHEAD;
#pragma unroll
      for (int dt = 0; dt < 4; ++dt)
        Ob[rowbase + dt * 16 + colc] = (bf16_t)(o[dt][r] * inv);
    }
  }
}

// ---------------------------------------------------------------------------
// Kernel 3: output projection. O(8192x1024 bf16) * Wproj(1024x1024 f32->bf16)
// -> out fp32. WG tile 128x128, wave tile 32x64 (2x4 WMMA).
// ---------------------------------------------------------------------------
__global__ __launch_bounds__(256)
void proj_gemm_kernel(const bf16_t* __restrict__ O, const float* __restrict__ W,
                      float* __restrict__ Out)
{
  __shared__ __align__(16) bf16_t Alds[128 * 40];
  __shared__ __align__(16) bf16_t Bt[128 * 40];

  const int m0   = blockIdx.x * 128;
  const int n0   = blockIdx.y * 128;
  const int tid  = threadIdx.x;
  const int wave = tid >> 5;
  const int wm   = (wave >> 1) * 32;
  const int wn   = (wave & 1) * 64;

  v8f acc[2][4] = {};

  for (int k0 = 0; k0 < CDIM; k0 += 32) {
    __syncthreads();
#pragma unroll
    for (int i = 0; i < 2; ++i) {                      // A: bf16 copy, v8bf
      int c = tid + i * 256;                           // 512 chunks of 8 halfs
      int row = c >> 2, chunk = (c & 3) * 8;
      *(v8bf*)(Alds + row * 40 + chunk) =
          *(const v8bf*)(O + (size_t)(m0 + row) * CDIM + k0 + chunk);
    }
#pragma unroll
    for (int i = 0; i < 4; ++i) {                      // B: f32 -> bf16, transposed
      int c = tid + i * 256;                           // 1024 float4 chunks
      int krow = c >> 5, col4 = (c & 31) * 4;
      v4f wv = *(const v4f*)(W + (size_t)(k0 + krow) * CDIM + n0 + col4);
#pragma unroll
      for (int j = 0; j < 4; ++j) Bt[(col4 + j) * 40 + krow] = (bf16_t)wv[j];
    }
    if (k0 + 32 < CDIM) {
      __builtin_prefetch(O + (size_t)(m0 + (tid >> 1)) * CDIM + k0 + 32 + (tid & 1) * 16, 0, 0);
      __builtin_prefetch(W + (size_t)(k0 + 32 + (tid >> 5)) * CDIM + n0 + (tid & 31) * 4, 0, 0);
    }
    __syncthreads();

#pragma unroll
    for (int mi = 0; mi < 2; ++mi) {
      v16bf a = load_a_frag(Alds + (wm + mi * 16) * 40, 40);
#pragma unroll
      for (int ni = 0; ni < 4; ++ni) {
        v16bf b = load_b_frag_cm(Bt + (wn + ni * 16) * 40, 40);
        acc[mi][ni] = wmma_bf16(a, b, acc[mi][ni]);
      }
    }
  }

  const int lane  = tid & 31;
  const int col   = lane & 15;
  const int rbase = (lane >> 4) * 8;
#pragma unroll
  for (int mi = 0; mi < 2; ++mi)
#pragma unroll
    for (int ni = 0; ni < 4; ++ni)
#pragma unroll
      for (int r = 0; r < 8; ++r) {
        int m = m0 + wm + mi * 16 + rbase + r;
        int n = n0 + wn + ni * 16 + col;
        Out[(size_t)m * CDIM + n] = acc[mi][ni][r];
      }
}

// ---------------------------------------------------------------------------
extern "C" void kernel_launch(void* const* d_in, const int* in_sizes, int n_in,
                              void* d_out, int out_size, void* d_ws, size_t ws_size,
                              hipStream_t stream) {
  (void)in_sizes; (void)n_in; (void)out_size; (void)ws_size;
  const float* x     = (const float*)d_in[0];   // (4,2048,1024) f32
  const float* Wkqv  = (const float*)d_in[1];   // (1024,3072)   f32
  const float* Wproj = (const float*)d_in[2];   // (1024,1024)   f32
  float* out = (float*)d_out;                   // (4,2048,1024) f32

  // Workspace: K,Q,V,O as bf16 [B][H][T][D] / [B][T][C]; 4 x 16 MiB = 64 MiB.
  const size_t headElems = (size_t)BATCH * HEADS * T_SEQ * DHEAD;  // 8Mi
  bf16_t* Kb = (bf16_t*)d_ws;
  bf16_t* Qb = Kb + headElems;
  bf16_t* Vb = Qb + headElems;
  bf16_t* Ob = Vb + headElems;

  qkv_gemm_kernel<<<dim3(64, 24), 256, 0, stream>>>(x, Wkqv, Kb, Qb, Vb);
  attn_kernel<<<dim3(T_SEQ / 128, HEADS, BATCH), 256, 0, stream>>>(Qb, Kb, Vb, Ob);
  proj_gemm_kernel<<<dim3(64, 8), 256, 0, stream>>>(Ob, Wproj, out);
}
